// ProtoNet_35373350650479
// MI455X (gfx1250) — matrix operand order repaired and checked by reading, not compiled, not run on hardware
//
#include <hip/hip_runtime.h>
#include <hip/hip_bf16.h>

typedef __attribute__((ext_vector_type(16))) _Float16 v16h;
typedef __attribute__((ext_vector_type(8)))  float    v8f;

#define NUM_CLASSES 64
#define FDIM 256
#define EPS_F 1e-8f

// ---------------------------------------------------------------- zero scratch
__global__ void k_zero(float* __restrict__ p, int n) {
    int i = blockIdx.x * blockDim.x + threadIdx.x;
    if (i < n) p[i] = 0.0f;
}

// ------------------------------------------------- segment-sum via LDS atomics
// grid = (NB, 2); blockIdx.y selects feature half [0,128) or [128,256).
// 64 classes * 128 dims * fp32 = 32 KB LDS accumulator per block.
__global__ __launch_bounds__(128)
void k_protosum(const float* __restrict__ sup, const int* __restrict__ lab,
                float* __restrict__ protoSum, float* __restrict__ counts,
                int rowsPerBlock) {
    __shared__ float acc[NUM_CLASSES * 128];
    __shared__ float cnt[NUM_CLASSES];
    const int t  = threadIdx.x;          // dim within half
    const int by = blockIdx.y;           // which feature half
    for (int i = t; i < NUM_CLASSES * 128; i += 128) acc[i] = 0.0f;
    if (by == 0 && t < NUM_CLASSES) cnt[t] = 0.0f;
    __syncthreads();

    const int r0 = blockIdx.x * rowsPerBlock;
    const float* base = sup + (size_t)r0 * FDIM + by * 128;
    for (int r = 0; r < rowsPerBlock; ++r) {
        const int l = lab[r0 + r];                      // uniform -> scalar load
        const float v = base[(size_t)r * FDIM + t];
        atomicAdd(&acc[l * 128 + t], v);                // ds_add_f32, bank-clean
        if (by == 0 && t == 0) atomicAdd(&cnt[l], 1.0f);
    }
    __syncthreads();

    for (int i = t; i < NUM_CLASSES * 128; i += 128) {
        const int c = i >> 7, d = i & 127;
        atomicAdd(&protoSum[c * FDIM + by * 128 + d], acc[i]); // global_atomic_add_f32
    }
    if (by == 0 && t < NUM_CLASSES) atomicAdd(&counts[t], cnt[t]);
}

// ------------- finalize prototypes: mean, fold 1/||p||, pack f16 B fragments
// B layout (16-bit 32x16, from CDNA5 ISA): N = lane&15,
// K(chunk-local) = (j&7) + 8*(lane>=16) + 16*(j>=8), j = half index 0..15 within v16h.
__global__ __launch_bounds__(256)
void k_finalize(const float* __restrict__ protoSum, const float* __restrict__ counts,
                _Float16* __restrict__ frags) {
    __shared__ float red[256];
    const int c = blockIdx.x;            // class
    const int t = threadIdx.x;           // feature k
    const float cn = fmaxf(counts[c], 1.0f);
    const float pv = protoSum[c * FDIM + t] / cn;
    red[t] = pv * pv;
    __syncthreads();
    for (int s = 128; s > 0; s >>= 1) { if (t < s) red[t] += red[t + s]; __syncthreads(); }
    const float rn = 1.0f / fmaxf(sqrtf(red[0]), EPS_F);

    const int kt = t >> 5;               // K chunk of 32
    const int kk = t & 31;               // K within chunk
    const int hi = (kk >> 3) & 1;        // lanes>=16 hold K 8..15, 24..31
    const int j  = (kk & 7) + ((kk >> 4) << 3);
    const int nt   = c >> 4;             // class tile (N tile)
    const int lane = (c & 15) + hi * 16;
    frags[(((kt * 4 + nt) * 32) + lane) * 16 + j] = (_Float16)(pv * rn);
}

// ----------------------- fused GEMM + normalize + log-softmax (1 wave = 16 q)
__global__ __launch_bounds__(128)
void k_gemm_lsm(const float* __restrict__ q, const _Float16* __restrict__ frags,
                float* __restrict__ out) {
    const int lane  = threadIdx.x & 31;
    const int wave  = threadIdx.x >> 5;
    const int tile  = blockIdx.x * 4 + wave;
    const int qbase = tile * 16;
    const int m     = lane & 15;
    const int hi    = (lane >> 4) & 1;
    const int koffs = hi * 8;            // A layout: lanes>=16 start at K+8
    const float* qrow = q + (size_t)(qbase + m) * FDIM;
    const v16h* Bf = (const v16h*)frags;

    v8f acc0 = {}, acc1 = {}, acc2 = {}, acc3 = {};
    float ssq = 0.0f;
#pragma unroll
    for (int kt = 0; kt < 8; ++kt) {
        const int k0 = kt * 32 + koffs;
        const float4 x0 = *(const float4*)(qrow + k0);
        const float4 x1 = *(const float4*)(qrow + k0 + 4);
        const float4 y0 = *(const float4*)(qrow + k0 + 16);
        const float4 y1 = *(const float4*)(qrow + k0 + 20);
        ssq += x0.x*x0.x + x0.y*x0.y + x0.z*x0.z + x0.w*x0.w
             + x1.x*x1.x + x1.y*x1.y + x1.z*x1.z + x1.w*x1.w
             + y0.x*y0.x + y0.y*y0.y + y0.z*y0.z + y0.w*y0.w
             + y1.x*y1.x + y1.y*y1.y + y1.z*y1.z + y1.w*y1.w;
        v16h a;
        a[0]=(_Float16)x0.x;  a[1]=(_Float16)x0.y;  a[2]=(_Float16)x0.z;  a[3]=(_Float16)x0.w;
        a[4]=(_Float16)x1.x;  a[5]=(_Float16)x1.y;  a[6]=(_Float16)x1.z;  a[7]=(_Float16)x1.w;
        a[8]=(_Float16)y0.x;  a[9]=(_Float16)y0.y;  a[10]=(_Float16)y0.z; a[11]=(_Float16)y0.w;
        a[12]=(_Float16)y1.x; a[13]=(_Float16)y1.y; a[14]=(_Float16)y1.z; a[15]=(_Float16)y1.w;
        const v16h b0 = Bf[(kt*4 + 0)*32 + lane];
        const v16h b1 = Bf[(kt*4 + 1)*32 + lane];
        const v16h b2 = Bf[(kt*4 + 2)*32 + lane];
        const v16h b3 = Bf[(kt*4 + 3)*32 + lane];
        acc0 = __builtin_amdgcn_wmma_f32_16x16x32_f16(false, a, false, b0, (short)0, acc0, false, false);
        acc1 = __builtin_amdgcn_wmma_f32_16x16x32_f16(false, a, false, b1, (short)0, acc1, false, false);
        acc2 = __builtin_amdgcn_wmma_f32_16x16x32_f16(false, a, false, b2, (short)0, acc2, false, false);
        acc3 = __builtin_amdgcn_wmma_f32_16x16x32_f16(false, a, false, b3, (short)0, acc3, false, false);
    }
    // lanes L and L+16 cover disjoint K halves of row (L&15): combine
    ssq += __shfl_xor(ssq, 16, 32);
    const float inv = 1.0f / fmaxf(sqrtf(ssq), EPS_F);

#pragma unroll
    for (int r = 0; r < 8; ++r) {
        const int M = r + hi * 8;                  // C/D layout row for VGPR r
        const float iq = __shfl(inv, M, 32);       // lane M holds row M's 1/norm
        float s0 = acc0[r] * iq;
        float s1 = acc1[r] * iq;
        float s2 = acc2[r] * iq;
        float s3 = acc3[r] * iq;
        float mx = fmaxf(fmaxf(s0, s1), fmaxf(s2, s3));
#pragma unroll
        for (int msk = 1; msk < 16; msk <<= 1) mx = fmaxf(mx, __shfl_xor(mx, msk, 32));
        float e = __expf(s0 - mx) + __expf(s1 - mx) + __expf(s2 - mx) + __expf(s3 - mx);
#pragma unroll
        for (int msk = 1; msk < 16; msk <<= 1) e += __shfl_xor(e, msk, 32);
        const float lse = mx + __logf(e);
        float* o = out + (size_t)(qbase + M) * NUM_CLASSES + (lane & 15);
        o[0]  = s0 - lse;
        o[16] = s1 - lse;
        o[32] = s2 - lse;
        o[48] = s3 - lse;
    }
}

// ---------------------------------------------------------------------- launch
extern "C" void kernel_launch(void* const* d_in, const int* in_sizes, int n_in,
                              void* d_out, int out_size, void* d_ws, size_t ws_size,
                              hipStream_t stream) {
    const float* support = (const float*)d_in[0];
    const int*   labels  = (const int*)d_in[1];
    const float* query   = (const float*)d_in[2];
    float* out = (float*)d_out;

    char* ws = (char*)d_ws;
    float*     protoSum = (float*)ws;               // 64*256 f32 = 65536 B
    float*     counts   = (float*)(ws + 65536);     // 64 f32
    _Float16*  frags    = (_Float16*)(ws + 66560);  // 8*4*32*16 f16 = 32768 B

    const int nrows = in_sizes[0] / FDIM;           // 262144
    const int nq    = in_sizes[2] / FDIM;           // 131072

    // 1. zero the accumulation scratch (protoSum + counts are contiguous)
    const int nz = NUM_CLASSES * FDIM + NUM_CLASSES;
    k_zero<<<(nz + 255) / 256, 256, 0, stream>>>(protoSum, nz);

    // 2. segment sum
    const int NB = 512;
    k_protosum<<<dim3(NB, 2), dim3(128), 0, stream>>>(support, labels, protoSum,
                                                      counts, nrows / NB);

    // 3. finalize prototypes -> packed f16 WMMA B fragments
    k_finalize<<<NUM_CLASSES, 256, 0, stream>>>(protoSum, counts, frags);

    // 4. fused GEMM + log-softmax: 16 queries/wave, 4 waves/block
    k_gemm_lsm<<<nq / 64, 128, 0, stream>>>(query, frags, out);
}
